// OutputPPBlock_7816840478892
// MI455X (gfx1250) — matrix operand order repaired and checked by reading, not compiled, not run on hardware
//
#include <hip/hip_runtime.h>

typedef __attribute__((ext_vector_type(16))) __bf16 v16bf;
typedef __attribute__((ext_vector_type(8)))  __bf16 bf16x8;
typedef __attribute__((ext_vector_type(8)))  float  v8f;

#define NN_  50000
#define E_   1000000
#define R_   6
#define H_   128
#define D_   256
#define L_   3

__device__ __forceinline__ unsigned short f32_to_bf16_rne(float f) {
  unsigned int u = __float_as_uint(f);
  unsigned int r = u + 0x7FFFu + ((u >> 16) & 1u);
  return (unsigned short)(r >> 16);
}

__device__ __forceinline__ float bf16_bits_to_f32(unsigned short b) {
  return __uint_as_float(((unsigned int)b) << 16);
}

// silu via fast hardware reciprocal (v_rcp_f32): output is rounded to bf16
// anyway, so the ~1 ulp rcp is plenty; avoids the IEEE div expansion
// (v_div_scale/v_div_fmas Newton sequence, ~13 VALU ops per value).
__device__ __forceinline__ float silu_fast(float v) {
  return v * __builtin_amdgcn_rcpf(1.0f + __expf(-v));
}

// ---------------------------------------------------------------- utilities
__global__ void zero_kernel(float* __restrict__ p, long n) {
  long i = (long)blockIdx.x * blockDim.x + threadIdx.x;
  long stride = (long)gridDim.x * blockDim.x;
  for (; i < n; i += stride) p[i] = 0.0f;
}

__global__ void cvt_bf16_kernel(const float* __restrict__ src,
                                unsigned short* __restrict__ dst, long n) {
  long i = (long)blockIdx.x * blockDim.x + threadIdx.x;
  long stride = (long)gridDim.x * blockDim.x;
  for (; i < n; i += stride) dst[i] = f32_to_bf16_rne(src[i]);
}

// ------------------------------------------- stage 1: edge MLP + scatter-add
// e[e,h] = (sum_r rbf[e,r] * W_rbf[h,r]) * x[e,h];  nodes[i[e],h] += e[e,h]
// Bandwidth bound (512 MB of x). block = 128 threads = H channels.
__global__ __launch_bounds__(128) void edge_scatter_kernel(
    const float* __restrict__ x, const float* __restrict__ rbf,
    const int* __restrict__ idx, const float* __restrict__ Wrbf,
    float* __restrict__ nodes) {
  const int h = threadIdx.x;  // 0..127
  const float w0 = Wrbf[h * R_ + 0], w1 = Wrbf[h * R_ + 1], w2 = Wrbf[h * R_ + 2];
  const float w3 = Wrbf[h * R_ + 3], w4 = Wrbf[h * R_ + 4], w5 = Wrbf[h * R_ + 5];
  for (int e = blockIdx.x; e < E_; e += gridDim.x) {
    const float* rb = rbf + (long)e * R_;
    float dot = w0 * rb[0] + w1 * rb[1] + w2 * rb[2] +
                w3 * rb[3] + w4 * rb[4] + w5 * rb[5];
    float val = dot * x[(long)e * H_ + h];
    atomicAdd(nodes + (long)idx[e] * H_ + h, val);
  }
}

// ------------------------------------------------- stage 2: WMMA bf16 GEMMs
// Out[m, n] = act( sum_k A[m,k] * W[n,k] + bias[n] )   (N fixed = 256)
// A: [M x K] bf16 row-major.  W: [256 x K] bf16 row-major (= B^T, matching
// the reference weight layout; this code's B fragments use contiguous K: lane n
// holds 16 contiguous K values of column n).
//
// Register blocking: each wave owns two 16-col N tiles and PRELOADS its B
// fragments for the entire K extent into registers (K/32 x 2 x v16bf), then
// sweeps MT M-tiles, so the steady-state loop is 2 A-loads : 2 WMMAs and the
// weight matrix is fetched once per block instead of once per M-tile.
template <int K, int MT>
__global__ __launch_bounds__(256) void gemm_bias_act_kernel(
    const unsigned short* __restrict__ A, const unsigned short* __restrict__ W,
    const float* __restrict__ bias, unsigned short* __restrict__ Out, int act) {
  constexpr int KT = K / 32;
  const int lane = threadIdx.x & 31;
  const int wave = threadIdx.x >> 5;
  const long mBase = (long)blockIdx.x * (16 * MT);
  const int n0 = wave * 32;            // this wave's two N tiles: n0, n0+16
  const int nl = lane & 15;
  const int hi = lane >> 4;            // lane half
  const int akb = hi ? 16 : 0;         // A-frag K sub-offset (contiguous, matches B)
  const int bkb = hi ? 16 : 0;         // B-frag K sub-offset (contiguous, matches A)

  const __bf16* Ab = (const __bf16*)A;
  const __bf16* Wb = (const __bf16*)W;
  const __bf16* w0row = Wb + (long)(n0 + nl) * K;
  const __bf16* w1row = Wb + (long)(n0 + 16 + nl) * K;

  // Preload all B fragments into registers (fully unrolled => no scratch).
  v16bf b0[KT], b1[KT];
#pragma unroll
  for (int t = 0; t < KT; ++t) {
    b0[t] = *(const v16bf*)(w0row + t * 32 + bkb);
    b1[t] = *(const v16bf*)(w1row + t * 32 + bkb);
  }

  const float bias0 = bias[n0 + nl];
  const float bias1 = bias[n0 + 16 + nl];

  for (int mt = 0; mt < MT; ++mt) {
    const long mTile = mBase + mt * 16;
    const __bf16* arow = Ab + (mTile + nl) * (long)K;
    v8f c0 = {}; v8f c1 = {};
#pragma unroll
    for (int t = 0; t < KT; ++t) {
      union { v16bf v; bf16x8 h[2]; } af;
      af.h[0] = *(const bf16x8*)(arow + t * 32 + akb);       // K sub 0..7
      af.h[1] = *(const bf16x8*)(arow + t * 32 + 8 + akb);   // K sub 8..15
      c0 = __builtin_amdgcn_wmma_f32_16x16x32_bf16(false, af.v, false, b0[t],
                                                   (short)0, c0, false, false);
      c1 = __builtin_amdgcn_wmma_f32_16x16x32_bf16(false, af.v, false, b1[t],
                                                   (short)0, c1, false, false);
    }
    // Epilogue. C/D layout: VGPR r, lanes 0-15 -> (m=r, n=lane);
    //                               lanes 16-31 -> (m=r+8, n=lane-16).
#pragma unroll
    for (int r = 0; r < 8; ++r) {
      const long m = mTile + r + 8 * hi;
      float v0 = c0[r] + bias0;
      float v1 = c1[r] + bias1;
      if (act) {
        v0 = silu_fast(v0);
        v1 = silu_fast(v1);
      }
      Out[m * D_ + n0 + nl]      = f32_to_bf16_rne(v0);
      Out[m * D_ + n0 + 16 + nl] = f32_to_bf16_rne(v1);
    }
  }
}

// ----------------------------------------- stage 3: final projection (O = 1)
// out[m] = sum_d h[m,d] * W_out[d].  One wave per row, butterfly reduction.
__global__ __launch_bounds__(256) void out_proj_kernel(
    const unsigned short* __restrict__ Hb, const float* __restrict__ Wout,
    float* __restrict__ out, int M) {
  const int lane = threadIdx.x & 31;
  const int wave = threadIdx.x >> 5;
  const long m = (long)blockIdx.x * 8 + wave;
  if (m >= M) return;
  const unsigned short* hrow = Hb + m * D_;
  float s = 0.0f;
#pragma unroll
  for (int j = 0; j < D_ / 32; ++j)
    s += bf16_bits_to_f32(hrow[lane + 32 * j]) * Wout[lane + 32 * j];
#pragma unroll
  for (int off = 16; off > 0; off >>= 1) s += __shfl_xor(s, off, 32);
  if (lane == 0) out[m] = s;
}

// --------------------------------------------------------------------- host
extern "C" void kernel_launch(void* const* d_in, const int* in_sizes, int n_in,
                              void* d_out, int out_size, void* d_ws, size_t ws_size,
                              hipStream_t stream) {
  const float* x    = (const float*)d_in[0];
  const float* rbf  = (const float*)d_in[1];
  const int*   idx  = (const int*)d_in[2];
  /* d_in[3] = num_nodes scalar (NN_ hardcoded) */
  const float* Wrbf = (const float*)d_in[4];
  const float* Wup  = (const float*)d_in[5];
  const float* bup  = (const float*)d_in[6];
  const float* Ws   = (const float*)d_in[7];
  const float* bs   = (const float*)d_in[8];
  const float* Wout = (const float*)d_in[9];
  float* out = (float*)d_out;

  // Workspace layout (all offsets 256B aligned; total 64,458,752 bytes).
  // n_f32 region is dead after cvt -> reused as first bf16 activation buffer.
  char* ws = (char*)d_ws;
  float*          n_f32  = (float*)(ws + 0);                  // 25,600,000 B
  unsigned short* hA     = (unsigned short*)(ws + 0);         // aliases n_f32
  unsigned short* n_b16  = (unsigned short*)(ws + 25600000);  // 12,800,000 B
  unsigned short* hB     = (unsigned short*)(ws + 38400000);  // 25,600,000 B
  unsigned short* Wup_b  = (unsigned short*)(ws + 64000000);  //     65,536 B
  unsigned short* Ws_b   = (unsigned short*)(ws + 64065536);  //    393,216 B

  // 1. zero node accumulators, scatter edge contributions
  zero_kernel<<<2048, 256, 0, stream>>>(n_f32, (long)NN_ * H_);
  edge_scatter_kernel<<<4096, 128, 0, stream>>>(x, rbf, idx, Wrbf, n_f32);

  // 2. convert node features + weights to bf16
  cvt_bf16_kernel<<<2048, 256, 0, stream>>>(n_f32, n_b16, (long)NN_ * H_);
  cvt_bf16_kernel<<<64, 256, 0, stream>>>(Wup, Wup_b, (long)D_ * H_);
  cvt_bf16_kernel<<<192, 256, 0, stream>>>(Ws, Ws_b, (long)L_ * D_ * D_);

  // 3. up-projection: [NN,128] x [128,256] + b_up (no activation)
  //    50000 = 16 * 5 * 625  ->  MT=5 M-tiles per block, 625 blocks.
  gemm_bias_act_kernel<H_, 5><<<625, 256, 0, stream>>>(n_b16, Wup_b, bup, hA, 0);

  // 4. three silu layers: [NN,256] x [256,256]
  unsigned short* a = hA;
  unsigned short* b = hB;
  for (int l = 0; l < L_; ++l) {
    gemm_bias_act_kernel<D_, 5><<<625, 256, 0, stream>>>(
        a, Ws_b + (long)l * D_ * D_, bs + (long)l * D_, b, 1);
    unsigned short* t = a; a = b; b = t;
  }

  // 5. final projection to O=1
  out_proj_kernel<<<(NN_ + 7) / 8, 256, 0, stream>>>(a, Wout, out, NN_);
}
